// SoftDTWDivergence_68753836474667
// MI455X (gfx1250) — matrix operand order repaired and checked by reading, not compiled, not run on hardware
//
#include <hip/hip_runtime.h>

typedef _Float16 h16;
typedef __attribute__((ext_vector_type(8)))  _Float16 v8h;
typedef __attribute__((ext_vector_type(16))) _Float16 v16h;
typedef __attribute__((ext_vector_type(8)))  float    v8f;

#define BIGV 1e30f
static constexpr int Bb = 8, Tt = 512, Dd = 128;

// ---------------------------------------------------------------------------
// Kernel 1: row-normalize x,y; emit hi/lo f16 splits in row-major (A operand)
// and transposed D x T layouts (unused; B also reads row-major). One wave per row (128 elems).
// ---------------------------------------------------------------------------
__global__ __launch_bounds__(256) void norm_split(
    const float* __restrict__ x, const float* __restrict__ y,
    h16* __restrict__ XhiR, h16* __restrict__ XloR,
    h16* __restrict__ YhiR, h16* __restrict__ YloR,
    h16* __restrict__ XhiT, h16* __restrict__ XloT,
    h16* __restrict__ YhiT, h16* __restrict__ YloT)
{
    int gw   = (blockIdx.x * blockDim.x + threadIdx.x) >> 5;  // row id 0..8191
    int lane = threadIdx.x & 31;
    bool isY = gw >= (Bb * Tt);
    int row  = isY ? gw - Bb * Tt : gw;                       // 0..4095
    const float* src = (isY ? y : x) + (size_t)row * Dd;

    float4 v = ((const float4*)src)[lane];
    float ss = v.x * v.x + v.y * v.y + v.z * v.z + v.w * v.w;
    #pragma unroll
    for (int m = 16; m >= 1; m >>= 1) ss += __shfl_xor(ss, m, 32);
    float inv = 1.0f / fmaxf(sqrtf(ss), 1e-12f);

    h16* hiR = isY ? YhiR : XhiR;
    h16* loR = isY ? YloR : XloR;
    h16* hiT = isY ? YhiT : XhiT;
    h16* loT = isY ? YloT : XloT;

    int b = row >> 9, t = row & 511;
    float vals[4] = {v.x * inv, v.y * inv, v.z * inv, v.w * inv};
    #pragma unroll
    for (int c = 0; c < 4; ++c) {
        float f  = vals[c];
        h16 hi = (h16)f;
        h16 lo = (h16)(f - (float)hi);
        int d = lane * 4 + c;
        hiR[(size_t)row * Dd + d] = hi;
        loR[(size_t)row * Dd + d] = lo;
        hiT[(size_t)b * Dd * Tt + (size_t)d * Tt + t] = hi;
        loT[(size_t)b * Dd * Tt + (size_t)d * Tt + t] = lo;
    }
}

// ---------------------------------------------------------------------------
// Kernel 2: cost matrices D[m] = 1 - Pn * Qn^T via f16 hi/lo split WMMA.
// One wave computes one 16x16 tile; 4 K-steps x 3 precision terms = 12 WMMAs.
// m = type*8 + b with type 0:xy 1:xx 2:yy.
// ---------------------------------------------------------------------------
__device__ __forceinline__ v16h cat8(v8h a, v8h b) {
    return __builtin_shufflevector(a, b, 0,1,2,3,4,5,6,7,8,9,10,11,12,13,14,15);
}

__global__ __launch_bounds__(256) void cost_gemm(
    const h16* __restrict__ XhiR, const h16* __restrict__ XloR,
    const h16* __restrict__ YhiR, const h16* __restrict__ YloR,
    const h16* __restrict__ XhiT, const h16* __restrict__ XloT,
    const h16* __restrict__ YhiT, const h16* __restrict__ YloT,
    float* __restrict__ Dmats)
{
    int lane = threadIdx.x & 31;
    int gw   = (blockIdx.x * blockDim.x + threadIdx.x) >> 5;
    int m    = gw >> 10;          // matrix 0..23
    int tile = gw & 1023;
    int ti = tile >> 5, tj = tile & 31;
    int type = m >> 3, b = m & 7;

    const h16* PhiR = (type == 2) ? YhiR : XhiR;
    const h16* PloR = (type == 2) ? YloR : XloR;
    const h16* QhiT = (type == 1) ? XhiR : YhiR;
    const h16* QloT = (type == 1) ? XloR : YloR;

    int r = lane & 15, s = lane >> 4;   // ISA A layout: lanes 16-31 take K-base +8
    const h16* aHi = PhiR + ((size_t)b * Tt + ti * 16 + r) * Dd + 8 * s;
    const h16* aLo = PloR + ((size_t)b * Tt + ti * 16 + r) * Dd + 8 * s;
    const h16* bHi = QhiT + ((size_t)b * Tt + tj * 16 + r) * Dd + 8 * s;
    const h16* bLo = QloT + ((size_t)b * Tt + tj * 16 + r) * Dd + 8 * s;

    v8f acc = {};
    #pragma unroll
    for (int k0 = 0; k0 < Dd; k0 += 32) {
        v16h Ah = cat8(*(const v8h*)(aHi + k0), *(const v8h*)(aHi + k0 + 16));
        v16h Al = cat8(*(const v8h*)(aLo + k0), *(const v8h*)(aLo + k0 + 16));
        v16h Bh = cat8(*(const v8h*)(bHi + k0), *(const v8h*)(bHi + k0 + 16));
        v16h Bl = cat8(*(const v8h*)(bLo + k0), *(const v8h*)(bLo + k0 + 16));
        acc = __builtin_amdgcn_wmma_f32_16x16x32_f16(false, Ah, false, Bh, (short)0, acc, false, false);
        acc = __builtin_amdgcn_wmma_f32_16x16x32_f16(false, Ah, false, Bl, (short)0, acc, false, false);
        acc = __builtin_amdgcn_wmma_f32_16x16x32_f16(false, Al, false, Bh, (short)0, acc, false, false);
    }

    float* Dm = Dmats + (size_t)m * Tt * Tt;
    int col = tj * 16 + (lane & 15);
    #pragma unroll
    for (int reg = 0; reg < 8; ++reg) {
        int rowi = ti * 16 + reg + 8 * s;   // ISA C/D layout: lanes 16-31 -> M+8
        Dm[(size_t)rowi * Tt + col] = 1.0f - acc[reg];
    }
}

// ---------------------------------------------------------------------------
// Kernel 3: soft-DTW anti-diagonal DP. One block per matrix, 512 threads,
// 3 rotating LDS diagonals, one barrier per diagonal, cost load for the next
// diagonal issued before the barrier so L2 latency overlaps the sync.
// ---------------------------------------------------------------------------
__global__ __launch_bounds__(512) void sdtw_dp(const float* __restrict__ Dmats,
                                               float* __restrict__ res)
{
    __shared__ float bufA[513], bufB[513], bufC[513];
    const float* Dm = Dmats + (size_t)blockIdx.x * Tt * Tt;
    int tid = threadIdx.x;
    int i = tid + 1;                      // cell row index 1..512

    bufA[i] = BIGV; bufB[i] = BIGV;
    if (tid == 0) { bufA[0] = 0.0f; bufB[0] = BIGV; bufC[0] = BIGV; }
    __syncthreads();

    float *p2 = bufA, *p1 = bufB, *cu = bufC;

    // prefetch cost for k = 2
    int j = 2 - i;
    float cost = 0.0f;
    if (j >= 1 && j <= Tt) cost = Dm[(size_t)(i - 1) * Tt + (j - 1)];

    for (int k = 2; k <= 2 * Tt; ++k) {
        j = k - i;
        bool valid = (j >= 1) && (j <= Tt);
        float a = p2[i - 1];              // R[i-1, j-1]
        float bq = p1[i];                 // R[i,   j-1]
        float c = p1[i - 1];              // R[i-1, j  ]
        float mn = fminf(a, fminf(bq, c));
        float ssum = __expf(mn - a) + __expf(mn - bq) + __expf(mn - c);
        float nv = valid ? (cost + (mn - __logf(ssum))) : BIGV;

        // prefetch cost for k+1 (hides L2 latency under the barrier)
        int jn = (k + 1) - i;
        float costn = 0.0f;
        if (jn >= 1 && jn <= Tt) costn = Dm[(size_t)(i - 1) * Tt + (jn - 1)];

        cu[i] = nv;
        if (tid == 0) cu[0] = BIGV;       // border R[0, k] = inf
        __syncthreads();
        float* t = p2; p2 = p1; p1 = cu; cu = t;
        cost = costn;
    }
    if (tid == Tt - 1) res[blockIdx.x] = p1[Tt];   // R[n, m]
}

// ---------------------------------------------------------------------------
// Kernel 4: divergence combine (fixed order -> deterministic).
// ---------------------------------------------------------------------------
__global__ void combine(const float* __restrict__ res, float* __restrict__ out) {
    int b = threadIdx.x;
    if (b < Bb) out[b] = res[b] - 0.5f * (res[Bb + b] + res[2 * Bb + b]);
}

// ---------------------------------------------------------------------------
extern "C" void kernel_launch(void* const* d_in, const int* in_sizes, int n_in,
                              void* d_out, int out_size, void* d_ws, size_t ws_size,
                              hipStream_t stream) {
    (void)in_sizes; (void)n_in; (void)out_size; (void)ws_size;
    const float* x = (const float*)d_in[0];
    const float* y = (const float*)d_in[1];
    float* out = (float*)d_out;

    char* ws = (char*)d_ws;
    const size_t MB = 1ull << 20;
    h16* XhiR = (h16*)(ws + 0 * MB);
    h16* XloR = (h16*)(ws + 1 * MB);
    h16* YhiR = (h16*)(ws + 2 * MB);
    h16* YloR = (h16*)(ws + 3 * MB);
    h16* XhiT = (h16*)(ws + 4 * MB);
    h16* XloT = (h16*)(ws + 5 * MB);
    h16* YhiT = (h16*)(ws + 6 * MB);
    h16* YloT = (h16*)(ws + 7 * MB);
    float* Dmats = (float*)(ws + 8 * MB);                       // 24 * 512*512 * 4B
    float* res   = (float*)(ws + 8 * MB + 24ull * Tt * Tt * 4); // 24 floats

    // 8192 rows, one wave each -> 1024 blocks of 8 waves
    norm_split<<<1024, 256, 0, stream>>>(x, y, XhiR, XloR, YhiR, YloR,
                                         XhiT, XloT, YhiT, YloT);
    // 24 matrices * 1024 tiles, one wave each -> 3072 blocks of 8 waves
    cost_gemm<<<3072, 256, 0, stream>>>(XhiR, XloR, YhiR, YloR,
                                        XhiT, XloT, YhiT, YloT, Dmats);
    sdtw_dp<<<24, 512, 0, stream>>>(Dmats, res);
    combine<<<1, 32, 0, stream>>>(res, out);
}
